// CircumpunctSSMv2_35321811042499
// MI455X (gfx1250) — matrix operand order repaired and checked by reading, not compiled, not run on hardware
//
#include <hip/hip_runtime.h>
#include <math.h>

typedef _Float16 h4  __attribute__((ext_vector_type(4)));
typedef _Float16 h8  __attribute__((ext_vector_type(8)));
typedef _Float16 h16 __attribute__((ext_vector_type(16)));
typedef float    f8  __attribute__((ext_vector_type(8)));

#define LDS_AS __attribute__((address_space(3)))

#define BATCH 8
#define TSTEPS 2048
#define DMODEL 1024
#define NSTATE 64
#define MROWS (BATCH * TSTEPS)        // 16384
#define NCAT 464                       // 449 real columns padded to 29 tiles of 16
#define BPAD (DMODEL + 8)              // LDS row stride in halves: 2064B -> conflict-free

// ---------------------------------------------------------------- prep kernels

__global__ void cvt_f32_to_f16(const float* __restrict__ src, _Float16* __restrict__ dst, int n) {
  int i = blockIdx.x * blockDim.x + threadIdx.x;
  if (i < n) dst[i] = (_Float16)src[i];
}

// Pack concatenated projection weights, transposed to [N=464][K=1024] f16.
// Column map: [0,64) W_dt | [64,192) W_B | [192,320) W_x | {320} W_gamma | [321,449) W_res | pad 0
__global__ void pack_wcat(const float* __restrict__ Wdt, const float* __restrict__ WB,
                          const float* __restrict__ Wx,  const float* __restrict__ Wg,
                          const float* __restrict__ Wres, _Float16* __restrict__ Wc) {
  int idx = blockIdx.x * blockDim.x + threadIdx.x;
  if (idx >= NCAT * DMODEL) return;
  int j = idx >> 10;
  int k = idx & (DMODEL - 1);
  float v = 0.0f;
  if      (j < 64)  v = Wdt[k * 64 + j];
  else if (j < 192) v = WB[k * 128 + (j - 64)];
  else if (j < 320) v = Wx[k * 128 + (j - 192)];
  else if (j == 320) v = Wg[k];
  else if (j < 449) v = Wres[k * 128 + (j - 321)];
  Wc[idx] = (_Float16)v;
}

// W_out (128 x 1024) -> transposed f16 [N=1024][K=128]
__global__ void pack_wout(const float* __restrict__ Wout, _Float16* __restrict__ Wo) {
  int idx = blockIdx.x * blockDim.x + threadIdx.x;
  if (idx >= DMODEL * 128) return;
  int j = idx >> 7;
  int k = idx & 127;
  Wo[idx] = (_Float16)Wout[k * DMODEL + j];
}

// ---------------------------------------------------------------- WMMA GEMMs
// Fragment maps (16-bit):
//  A: M = lane%16; halves 0..7 = K[base..base+7], 8..15 = K[base+16..base+23], base=(lane>=16)*8
//  B: N = lane%16; halves 0..7 = K[base..base+7], 8..15 = K[base+16..base+23], base=(lane>=16)*8

// proj GEMM: 4 waves of a block share one B column tile (tn), staged into LDS
// via global_load_async_to_lds_b128 (ASYNCcnt path); each wave owns tm = tmBase+wid.
__global__ __launch_bounds__(128) void proj_gemm(const _Float16* __restrict__ Xh,
                                                 const _Float16* __restrict__ Wc,
                                                 float* __restrict__ DT, float* __restrict__ BT,
                                                 float* __restrict__ U,  float* __restrict__ G,
                                                 float* __restrict__ XC) {
  __shared__ _Float16 bs[16 * BPAD];          // 33 KB, padded rows
  const int K = DMODEL;
  const int NT = NCAT / 16;                   // 29
  int wid  = threadIdx.x >> 5;
  int lane = threadIdx.x & 31;
  int tn     = blockIdx.x % NT;
  int tmBase = (blockIdx.x / NT) * 4;
  int r = lane & 15, hi = lane >> 4;

  // ---- stage B tile (16 x 1024 halves) into LDS asynchronously ----
  unsigned lbase = (unsigned)(unsigned long long)(LDS_AS _Float16*)bs;
#pragma unroll
  for (int rr = 0; rr < 4; ++rr) {
    int row = wid * 4 + rr;
#pragma unroll
    for (int c = 0; c < 4; ++c) {
      const _Float16* g = Wc + (size_t)(tn * 16 + row) * K + c * 256 + lane * 8;
      unsigned loff = lbase + (unsigned)(row * (BPAD * 2) + c * 512 + lane * 16);
      asm volatile("global_load_async_to_lds_b128 %0, %1, off"
                   :: "v"(loff), "v"(g) : "memory");
    }
  }
#if __has_builtin(__builtin_amdgcn_s_wait_asynccnt)
  __builtin_amdgcn_s_wait_asynccnt(0);
#else
  asm volatile("s_wait_asynccnt 0x0" ::: "memory");
#endif
  __syncthreads();

  // ---- K loop: A from global (L2), B fragments from LDS ----
  int tm = tmBase + wid;
  const _Float16* a0 = Xh + (size_t)(tm * 16 + r) * K + hi * 8;
  const _Float16* b0 = bs + r * BPAD + hi * 8;
  f8 acc = {};
  for (int k = 0; k < K; k += 32) {
    h8 alo = *(const h8*)(a0 + k);
    h8 ahi = *(const h8*)(a0 + k + 16);
    h8 blo = *(const h8*)(b0 + k);
    h8 bhi = *(const h8*)(b0 + k + 16);
    h16 av = __builtin_shufflevector(alo, ahi, 0,1,2,3,4,5,6,7,8,9,10,11,12,13,14,15);
    h16 bv = __builtin_shufflevector(blo, bhi, 0,1,2,3,4,5,6,7,8,9,10,11,12,13,14,15);
    acc = __builtin_amdgcn_wmma_f32_16x16x32_f16(false, av, false, bv, (short)0, acc, false, false);
  }

  int col = tn * 16 + r;
  int rowbase = tm * 16 + hi * 8;
#pragma unroll
  for (int v = 0; v < 8; ++v) {
    int row = rowbase + v;
    float val = acc[v];
    if      (col < 64)  DT[(size_t)row * 64 + col] = val;
    else if (col < 192) BT[(size_t)row * 128 + (col - 64)] = val;
    else if (col < 320) U[(size_t)row * 128 + (col - 192)] = val;
    else if (col == 320) G[row] = val;
    else if (col < 449) XC[(size_t)row * 128 + (col - 321)] = val;
  }
}

__global__ __launch_bounds__(128) void out_gemm(const _Float16* __restrict__ Hh,
                                                const _Float16* __restrict__ Wo,
                                                float* __restrict__ Y) {
  const int K = 128;
  const int NT = DMODEL / 16;               // 64
  int wid  = threadIdx.x >> 5;
  int lane = threadIdx.x & 31;
  int tile = blockIdx.x * 4 + wid;
  int tm = tile / NT, tn = tile % NT;
  int r = lane & 15, hi = lane >> 4;

  const _Float16* a0 = Hh + (size_t)(tm * 16 + r) * K + hi * 8;
  const _Float16* b0 = Wo + (size_t)(tn * 16 + r) * K + hi * 8;
  f8 acc = {};
  for (int k = 0; k < K; k += 32) {
    h8 alo = *(const h8*)(a0 + k);
    h8 ahi = *(const h8*)(a0 + k + 16);
    h8 blo = *(const h8*)(b0 + k);
    h8 bhi = *(const h8*)(b0 + k + 16);
    h16 av = __builtin_shufflevector(alo, ahi, 0,1,2,3,4,5,6,7,8,9,10,11,12,13,14,15);
    h16 bv = __builtin_shufflevector(blo, bhi, 0,1,2,3,4,5,6,7,8,9,10,11,12,13,14,15);
    acc = __builtin_amdgcn_wmma_f32_16x16x32_f16(false, av, false, bv, (short)0, acc, false, false);
  }

  int col = tn * 16 + r;
  int rowbase = tm * 16 + hi * 8;
#pragma unroll
  for (int v = 0; v < 8; ++v) {
    Y[(size_t)(rowbase + v) * DMODEL + col] = acc[v];
  }
}

// ---------------------------------------------------------------- scan kernel
// Trig-free reformulation (identical math up to rounding):
//   cos(phase(z)) = (z.x+1e-10)/r, sin(phase(z)) = z.y/r, r = sqrt((z.x+1e-10)^2+z.y^2)
//   cos^2((xp-dp)/2) = (1 + xhat.dhat)/2
//   mag^e = exp(e*log(mag))
// All heavy ops are HW TRANS instructions (v_sqrt/v_rcp/v_exp/v_log).

__device__ __forceinline__ float frcp(float x)  { return __builtin_amdgcn_rcpf(x); }
__device__ __forceinline__ float fsqrt(float x) { return __builtin_amdgcn_sqrtf(x); }

__device__ __forceinline__ float2 unitv(float2 z) {
  float zx = z.x + 1e-10f;
  float ri = frcp(fsqrt(zx * zx + z.y * z.y));
  return make_float2(zx * ri, z.y * ri);
}

__device__ __forceinline__ float2 cvg(float2 z, float e) {
  float m = fsqrt(z.x * z.x + z.y * z.y + 1e-8f);
  float c = __expf(e * __logf(fmaxf(m, 1e-8f)));
  float2 u = unitv(z);
  return make_float2(c * u.x, c * u.y);
}

__device__ __forceinline__ float2 clamp_over(float2 d, float cap, float2* ov) {
  float m = fsqrt(d.x * d.x + d.y * d.y + 1e-8f);
  if (m > cap) {
    float2 u = unitv(d);
    float om = m - cap;
    *ov = make_float2(om * u.x, om * u.y);
    float sc = cap * frcp(m);
    return make_float2(d.x * sc, d.y * sc);
  }
  *ov = make_float2(0.f, 0.f);
  return d;
}

__device__ __forceinline__ void absorb_state(float2 rel, float bq, float aexp,
                                             float2& dfa, float2& dmi, float2& dde) {
  float2 cf = cvg(rel, aexp);
  cf.x = fminf(fmaxf(cf.x, -10.f), 10.f);
  cf.y = fminf(fmaxf(cf.y, -10.f), 10.f);
  dfa.x += 0.05f * bq * cf.x;  dfa.y += 0.05f * bq * cf.y;
  float2 o1; dfa = clamp_over(dfa, 10.f, &o1);
  float2 c2 = cvg(o1, 0.4444444444444444f);      // (2/3)^2
  dmi.x += 0.01f * bq * c2.x;  dmi.y += 0.01f * bq * c2.y;
  float2 o2; dmi = clamp_over(dmi, 15.f, &o2);
  float2 c3 = cvg(o2, 0.2962962962962963f);      // (2/3)^3
  dde.x += 0.002f * bq * c3.x; dde.y += 0.002f * bq * c3.y;
  float m = fsqrt(dde.x * dde.x + dde.y * dde.y + 1e-8f);
  if (m > 20.f) { float sc = 20.f * frcp(m); dde.x *= sc; dde.y *= sc; }
}

__device__ __forceinline__ float2 retrieve_state(float2 xc, float2 dfa, float2 dmi, float2 dde,
                                                 float w0, float w1, float w2) {
  float2 xu = unitv(xc);
  float2 s = make_float2(0.f, 0.f);
  auto lvl = [&](float2 d, float cap, float w) {
    float2 du = unitv(d);
    float tg = 0.5f * (1.f + xu.x * du.x + xu.y * du.y);   // cos^2((xp-dp)/2)
    float dm = fsqrt(d.x * d.x + d.y * d.y + 1e-8f);
    float f = w * fsqrt(fminf(fmaxf(dm, 1e-6f), cap)) * tg * frcp(dm + 1e-8f);
    s.x += f * d.x; s.y += f * d.y;
  };
  lvl(dfa, 10.f, w0); lvl(dmi, 15.f, w1); lvl(dde, 20.f, w2);
  return s;
}

__device__ __forceinline__ float softplus_f(float x) {
  return (x > 20.f) ? x : __logf(1.f + __expf(x));
}

// One batch per block, one wave32; lane owns states 2*lane and 2*lane+1.
// Software-pipelined: step t+1 operands load while step t computes.
__global__ __launch_bounds__(32) void scan_kernel(
    const float* __restrict__ DT, const float* __restrict__ BTb,
    const float* __restrict__ Ub, const float* __restrict__ Gb,
    const float* __restrict__ XCb, const float* __restrict__ A_log,
    const float* __restrict__ level_weight, _Float16* __restrict__ Hh) {
  int b = blockIdx.x;
  int lane = threadIdx.x;
  int n0 = lane * 2;

  float Aa[2] = { expf(A_log[n0]), expf(A_log[n0 + 1]) };

  float l0 = level_weight[0], l1 = level_weight[1], l2 = level_weight[2];
  float lm = fmaxf(l0, fmaxf(l1, l2));
  float e0 = expf(l0 - lm), e1 = expf(l1 - lm), e2 = expf(l2 - lm);
  float es = e0 + e1 + e2;
  float w0 = e0 / es, w1 = e1 / es, w2 = e2 / es;

  const float STEP = 6.283185307179586f / (float)NSTATE;
  float2 h[2], df[2], dm[2], dd[2];
#pragma unroll
  for (int i = 0; i < 2; ++i) {
    float p = STEP * (n0 + i);
    float c = cosf(p), s = sinf(p);
    h[i]  = make_float2(0.f, 0.f);
    df[i] = make_float2(0.01f * c, 0.01f * s);
    dm[i] = make_float2(0.005f * c, 0.005f * s);
    dd[i] = make_float2(0.001f * c, 0.001f * s);
  }
  float balance = 0.5f;

  size_t base = (size_t)b * TSTEPS;
  // pipeline registers (current / next)
  float2 dt_c = *(const float2*)(DT  + base * 64  + n0);
  float4 bt_c = *(const float4*)(BTb + base * 128 + 2 * n0);
  float4 u_c  = *(const float4*)(Ub  + base * 128 + 2 * n0);
  float4 xc_c = *(const float4*)(XCb + base * 128 + 2 * n0);
  float  g_c  = Gb[base];

  for (int t = 0; t < TSTEPS; ++t) {
    size_t row  = base + t;
    size_t nrow = base + ((t + 1 < TSTEPS) ? t + 1 : t);
    // issue next-step loads early (independent of the recurrence)
    float2 dt_n = *(const float2*)(DT  + nrow * 64  + n0);
    float4 bt_n = *(const float4*)(BTb + nrow * 128 + 2 * n0);
    float4 u_n  = *(const float4*)(Ub  + nrow * 128 + 2 * n0);
    float4 xc_n = *(const float4*)(XCb + nrow * 128 + 2 * n0);
    float  g_n  = Gb[nrow];
    // keep lines ~8 steps ahead warm in WGP$ (global_prefetch_b8)
    size_t prow = base + ((t + 8 < TSTEPS) ? t + 8 : TSTEPS - 1);
    __builtin_prefetch(BTb + prow * 128 + 2 * n0, 0, 3);
    __builtin_prefetch(Ub  + prow * 128 + 2 * n0, 0, 3);
    __builtin_prefetch(XCb + prow * 128 + 2 * n0, 0, 3);

    float bal  = fminf(fmaxf(balance, 0.01f), 0.99f);
    float aexp = (1.f + bal) / (2.f + bal);
    float bq   = fmaxf(1.f - 2.f * fabsf(bal - 0.5f), 0.1f);
    float gamma = frcp(1.f + __expf(-g_c));

    float  dts[2] = { dt_c.x, dt_c.y };
    float2 btl[2] = { make_float2(bt_c.x, bt_c.y), make_float2(bt_c.z, bt_c.w) };
    float2 ul[2]  = { make_float2(u_c.x,  u_c.y),  make_float2(u_c.z,  u_c.w)  };
    float2 xcl[2] = { make_float2(xc_c.x, xc_c.y), make_float2(xc_c.z, xc_c.w) };

    float convl = 0.f, emel = 0.f;
#pragma unroll
    for (int i = 0; i < 2; ++i) {
      float dt = softplus_f(dts[i]);
      float al = __expf(-dt * Aa[i]);
      float2 rel = make_float2((1.f - al) * h[i].x, (1.f - al) * h[i].y);
      convl += rel.x * rel.x + rel.y * rel.y + 1e-8f;

      absorb_state(rel, bq, aexp, df[i], dm[i], dd[i]);
      float2 sf = retrieve_state(xcl[i], df[i], dm[i], dd[i], w0, w1, w2);

      float2 u = make_float2(ul[i].x + sf.x, ul[i].y + sf.y);
      float2 bt = btl[i];
      h[i].x = al * h[i].x + gamma * (bt.x * u.x - bt.y * u.y);
      h[i].y = al * h[i].y + gamma * (bt.x * u.y + bt.y * u.x);
      emel += h[i].x * h[i].x + h[i].y * h[i].y + 1e-8f;
    }

    float cs = convl, esum = emel;
#pragma unroll
    for (int m = 16; m > 0; m >>= 1) {
      cs   += __shfl_xor(cs,   m, 32);
      esum += __shfl_xor(esum, m, 32);
    }
    balance = 0.99f * balance + 0.01f * cs / (cs + esum + 1e-8f);

    h4 hv;
    hv[0] = (_Float16)h[0].x; hv[1] = (_Float16)h[0].y;
    hv[2] = (_Float16)h[1].x; hv[3] = (_Float16)h[1].y;
    *(h4*)(Hh + row * 128 + 2 * n0) = hv;

    dt_c = dt_n; bt_c = bt_n; u_c = u_n; xc_c = xc_n; g_c = g_n;
  }
}

// ---------------------------------------------------------------- launcher

extern "C" void kernel_launch(void* const* d_in, const int* in_sizes, int n_in,
                              void* d_out, int out_size, void* d_ws, size_t ws_size,
                              hipStream_t stream) {
  const float* x       = (const float*)d_in[0];
  const float* W_dt    = (const float*)d_in[1];
  const float* W_B     = (const float*)d_in[2];
  const float* W_x     = (const float*)d_in[3];
  const float* W_gamma = (const float*)d_in[4];
  const float* W_res   = (const float*)d_in[5];
  const float* W_out   = (const float*)d_in[6];
  const float* level_w = (const float*)d_in[7];
  const float* A_log   = (const float*)d_in[8];
  float* Y = (float*)d_out;

  char* ws = (char*)d_ws;
  size_t off = 0;
  auto alloc = [&](size_t bytes) -> void* {
    void* p = ws + off;
    off += (bytes + 255) & ~(size_t)255;
    return p;
  };
  _Float16* Xh = (_Float16*)alloc((size_t)MROWS * DMODEL * 2);   // 32 MB
  _Float16* Wc = (_Float16*)alloc((size_t)NCAT * DMODEL * 2);    // 0.93 MB
  _Float16* Wo = (_Float16*)alloc((size_t)DMODEL * 128 * 2);     // 0.25 MB
  float* DT = (float*)alloc((size_t)MROWS * 64 * 4);             // 4 MB
  float* BT = (float*)alloc((size_t)MROWS * 128 * 4);            // 8 MB
  float* U  = (float*)alloc((size_t)MROWS * 128 * 4);            // 8 MB
  float* G  = (float*)alloc((size_t)MROWS * 4);                  // 64 KB
  float* XC = (float*)alloc((size_t)MROWS * 128 * 4);            // 8 MB
  _Float16* Hh = (_Float16*)alloc((size_t)MROWS * 128 * 2);      // 4 MB

  int nx = MROWS * DMODEL;
  cvt_f32_to_f16<<<(nx + 255) / 256, 256, 0, stream>>>(x, Xh, nx);

  int nwc = NCAT * DMODEL;
  pack_wcat<<<(nwc + 255) / 256, 256, 0, stream>>>(W_dt, W_B, W_x, W_gamma, W_res, Wc);

  int nwo = DMODEL * 128;
  pack_wout<<<(nwo + 255) / 256, 256, 0, stream>>>(W_out, Wo);

  // (MROWS/16)/4 * (NCAT/16) blocks; block = 4 waves sharing one B tile
  proj_gemm<<<(1024 / 4) * 29, 128, 0, stream>>>(Xh, Wc, DT, BT, U, G, XC);

  scan_kernel<<<BATCH, 32, 0, stream>>>(DT, BT, U, G, XC, A_log, level_w, Hh);

  // (MROWS/16)*(DMODEL/16) = 65536 tiles, 4 waves/block
  out_gemm<<<65536 / 4, 128, 0, stream>>>(Hh, Wo, Y);
}
